// KGTransformer_80762565034487
// MI455X (gfx1250) — compile-verified, matrix operand order checked by reading, not run on hardware
//
#include <hip/hip_runtime.h>
#include <hip/hip_bf16.h>
#include <math.h>

typedef __attribute__((ext_vector_type(16))) __bf16    v16bf;
typedef __attribute__((ext_vector_type(8)))  float     v8f;
typedef __attribute__((ext_vector_type(4)))  unsigned  v4u;
typedef __attribute__((ext_vector_type(8)))  int       v8i;
typedef __attribute__((ext_vector_type(4)))  int       v4i;

#define HID 128
#define NH 8
#define HS 16
#define TN 8
#define NR 8
#define NLAYER 2
#define LN_EPS 1e-5f
#define RSQRT_D 0.25f   // 1/sqrt(HS=16)

// K-index inside a 16-bit A/B WMMA fragment (16x32 bf16 layout, ISA 7.12.2):
// element pair p lives in VGPR p; VGPR0-3 cover K=0..15 (split by lane half),
// VGPR4-7 cover K=16..31.
__device__ __forceinline__ int kidx(int e, int half) {
  int p = e >> 1, o = e & 1;
  return ((p < 4) ? (2 * p) : (16 + 2 * (p - 4))) + 8 * half + o;
}

// TDM: stage a 16xHID f32 tile (row stride ldx) into LDS. Rows past nRows
// zero-fill via tensor OOB. One issue per wave; completion via TENSORcnt.
__device__ __forceinline__ void tdm_stage_tile(const float* X, int ldx,
                                               int base, int nRows,
                                               float* sX) {
  unsigned long long ga = (unsigned long long)(uintptr_t)(X + (size_t)base * ldx);
  unsigned lds_off = (unsigned)(uintptr_t)sX;   // low 32 bits = LDS byte offset
  v4u g0 = { 1u,                                 // count=1 (valid descriptor)
             lds_off,
             (unsigned)(ga & 0xFFFFFFFFu),
             (unsigned)((ga >> 32) & 0x01FFFFFFu) | 0x80000000u };  // type=2
  unsigned rem = (unsigned)(nRows - base);
  v8i g1 = { (int)(2u << 16),                    // data_size=4B
             (int)((unsigned)ldx << 16),         // tensor_dim0 lo16 @ [31:16]
             (int)((rem & 0xFFFFu) << 16),       // tensor_dim1 lo16 @ [31:16]
             (int)(((rem >> 16) & 0xFFFFu) | (128u << 16)),  // dim1 hi | tile_dim0=128
             16,                                 // tile_dim1 = 16 rows
             ldx,                                // tensor_dim0_stride (lo32)
             0, 0 };
  v4i gz4 = { 0, 0, 0, 0 };
  v8i gz8 = { 0, 0, 0, 0, 0, 0, 0, 0 };
  __builtin_amdgcn_tensor_load_to_lds(g0, g1, gz4, gz4, gz8, 0);
}

// ---------------- typed linear: Y[row] = X[row] @ W[ntype[row]] --------------
// One wave per 16-row tile. Per-type masked WMMA; each row stored exactly once.
// Wswz: bf16 weights pre-swizzled in fragment order [t][n][kt][lane][e].
__global__ void typed_gemm128(const float* __restrict__ X, int ldx,
                              const int* __restrict__ ntype,
                              const __bf16* __restrict__ Wswz,
                              float* __restrict__ Y, int ldy,
                              int nRows) {
  __shared__ float sX[16 * HID];
  __shared__ int   sType[16];
  __shared__ unsigned sMask;
  const int tid  = threadIdx.x;          // 0..31 (wave32)
  const int base = blockIdx.x * 16;

  tdm_stage_tile(X, ldx, base, nRows, sX);   // async DMA into LDS

  if (tid < 16) sType[tid] = (base + tid < nRows) ? ntype[base + tid] : -1;
  if (tid == 0) {
    unsigned m = 0;
    for (int r = 0; r < 16 && base + r < nRows; ++r) m |= 1u << ntype[base + r];
    sMask = m;
  }
  __builtin_amdgcn_s_wait_tensorcnt(0);
  __syncthreads();

  const int half = tid >> 4;
  const int ln15 = tid & 15;             // A: row M ; B/D: col N

  // Hoist this lane's 8 output-row types/valid flags into registers so the
  // masked D-store needs no LDS traffic inside the type loop.
  int  rowType[8];
  bool rowOk[8];
  #pragma unroll
  for (int v = 0; v < 8; ++v) {
    int m2 = v + 8 * half;
    rowType[v] = sType[m2];
    rowOk[v]   = (base + m2) < nRows;
  }

  v16bf afr[4];
  #pragma unroll
  for (int kt = 0; kt < 4; ++kt)
    #pragma unroll
    for (int e = 0; e < 16; ++e)
      afr[kt][e] = (__bf16)sX[ln15 * HID + kt * 32 + kidx(e, half)];

  const unsigned tmask = sMask;
  for (int t = 0; t < TN; ++t) {
    if (!((tmask >> t) & 1u)) continue;
    for (int n = 0; n < 8; ++n) {
      v8f acc = {0.f, 0.f, 0.f, 0.f, 0.f, 0.f, 0.f, 0.f};
      #pragma unroll
      for (int kt = 0; kt < 4; ++kt) {
        size_t fi = ((size_t)t * 8 + n) * 4 + kt;
        v16bf bfr = *(const v16bf*)(Wswz + fi * 512 + (size_t)tid * 16);
        acc = __builtin_amdgcn_wmma_f32_16x16x32_bf16(
            false, afr[kt], false, bfr, (short)0, acc, false, false);
      }
      #pragma unroll
      for (int v = 0; v < 8; ++v) {
        if (rowOk[v] && rowType[v] == t)
          Y[(size_t)(base + v + 8 * half) * ldy + n * 16 + ln15] = acc[v];
      }
    }
  }
}

// ---------------- final FF: Y = X @ Wff + bff --------------------------------
__global__ void gemm128_bias(const float* __restrict__ X,
                             const __bf16* __restrict__ Wswz,  // 32 frags
                             const float* __restrict__ bias,
                             float* __restrict__ Y, int nRows) {
  __shared__ float sX[16 * HID];
  const int tid  = threadIdx.x;
  const int base = blockIdx.x * 16;

  tdm_stage_tile(X, HID, base, nRows, sX);
  __builtin_amdgcn_s_wait_tensorcnt(0);
  __syncthreads();

  const int half = tid >> 4;
  const int ln15 = tid & 15;

  v16bf afr[4];
  #pragma unroll
  for (int kt = 0; kt < 4; ++kt)
    #pragma unroll
    for (int e = 0; e < 16; ++e)
      afr[kt][e] = (__bf16)sX[ln15 * HID + kt * 32 + kidx(e, half)];

  for (int n = 0; n < 8; ++n) {
    v8f acc = {0.f, 0.f, 0.f, 0.f, 0.f, 0.f, 0.f, 0.f};
    #pragma unroll
    for (int kt = 0; kt < 4; ++kt) {
      size_t fi = (size_t)n * 4 + kt;
      v16bf bfr = *(const v16bf*)(Wswz + fi * 512 + (size_t)tid * 16);
      acc = __builtin_amdgcn_wmma_f32_16x16x32_bf16(
          false, afr[kt], false, bfr, (short)0, acc, false, false);
    }
    float bb = bias[n * 16 + ln15];
    #pragma unroll
    for (int v = 0; v < 8; ++v) {
      int gr = base + v + 8 * half;
      if (gr < nRows) Y[(size_t)gr * HID + n * 16 + ln15] = acc[v] + bb;
    }
  }
}

// ---- one-time: convert f32 weights (T x 128 x 128) into bf16 fragments -----
// out layout: frag fi = ((t*8 + n)*4 + kt); per frag: [lane 0..31][e 0..15].
__global__ void swizzle_weights(const float* __restrict__ W,
                                __bf16* __restrict__ out) {
  const int f = blockIdx.x;             // t*32 + n*4 + kt
  const int t = f >> 5, rem = f & 31, n = rem >> 2, kt = rem & 3;
  const int lane = threadIdx.x;         // 32
  const int half = lane >> 4, ln15 = lane & 15;
  const float* Wt = W + (size_t)t * HID * 128;
  __bf16* o = out + (size_t)f * 512 + (size_t)lane * 16;
  #pragma unroll
  for (int e = 0; e < 16; ++e)
    o[e] = (__bf16)Wt[(size_t)(kt * 32 + kidx(e, half)) * 128 + n * 16 + ln15];
}

// ---------------- edge phase -------------------------------------------------
__device__ __forceinline__ unsigned fenc(float x) {
  unsigned b = __float_as_uint(x);
  return (b & 0x80000000u) ? ~b : (b | 0x80000000u);
}
__device__ __forceinline__ float fdec(unsigned u) {
  return __uint_as_float((u & 0x80000000u) ? (u & 0x7FFFFFFFu) : ~u);
}

__global__ void edge_score(const float* __restrict__ kqv,   // N x 384 (k|q|v)
                           const int* __restrict__ src, const int* __restrict__ dst,
                           const int* __restrict__ etype,
                           const float* __restrict__ Watt_l, // NH*NR*16*16
                           const float* __restrict__ pri_l,  // NH*NR
                           float* __restrict__ a,            // E x NH
                           unsigned* __restrict__ nmax,      // N x NH (ordered-enc)
                           int E) {
  __shared__ float sW[NH * NR * HS * HS];   // 64 KB
  for (int i = threadIdx.x; i < NH * NR * HS * HS; i += blockDim.x) sW[i] = Watt_l[i];
  __syncthreads();
  int e = blockIdx.x * blockDim.x + threadIdx.x;
  if (e >= E) return;
  int s = src[e], d = dst[e], r = etype[e];
  const float* kb = kqv + (size_t)s * 384;
  const float* qb = kqv + (size_t)d * 384 + 128;
  for (int h = 0; h < NH; ++h) {
    float kv[HS], qv[HS];
    #pragma unroll
    for (int i = 0; i < HS; ++i) { kv[i] = kb[h * HS + i]; qv[i] = qb[h * HS + i]; }
    const float* Wm = sW + (h * NR + r) * HS * HS;
    float acc = 0.f;
    #pragma unroll
    for (int o = 0; o < HS; ++o) {
      float kw = 0.f;
      #pragma unroll
      for (int i = 0; i < HS; ++i) kw += kv[i] * Wm[i * HS + o];
      acc += kw * qv[o];
    }
    float ae = acc * pri_l[h * NR + r] * RSQRT_D;
    a[(size_t)e * NH + h] = ae;
    atomicMax(&nmax[(size_t)d * NH + h], fenc(ae));
  }
}

__global__ void edge_exp(float* __restrict__ a, const int* __restrict__ dst,
                         const unsigned* __restrict__ nmax,
                         float* __restrict__ den, int E) {
  int e = blockIdx.x * blockDim.x + threadIdx.x;
  if (e >= E) return;
  int d = dst[e];
  #pragma unroll
  for (int h = 0; h < NH; ++h) {
    float mx = fdec(nmax[(size_t)d * NH + h]);
    float ex = __expf(a[(size_t)e * NH + h] - mx);
    a[(size_t)e * NH + h] = ex;
    atomicAdd(&den[(size_t)d * NH + h], ex);
  }
}

__global__ void edge_msg(const float* __restrict__ kqv,
                         const int* __restrict__ src, const int* __restrict__ dst,
                         const int* __restrict__ etype,
                         const float* __restrict__ Wmsg_l,  // NH*NR*16*16
                         const float* __restrict__ a,       // ex
                         const float* __restrict__ den,
                         float* __restrict__ agg,           // N x 128
                         int E) {
  __shared__ float sW[NH * NR * HS * HS];   // 64 KB
  for (int i = threadIdx.x; i < NH * NR * HS * HS; i += blockDim.x) sW[i] = Wmsg_l[i];
  __syncthreads();
  int e = blockIdx.x * blockDim.x + threadIdx.x;
  if (e >= E) return;
  int s = src[e], d = dst[e], r = etype[e];
  const float* vb = kqv + (size_t)s * 384 + 256;
  for (int h = 0; h < NH; ++h) {
    float alpha = a[(size_t)e * NH + h] / (den[(size_t)d * NH + h] + 1e-16f);
    float vv[HS];
    #pragma unroll
    for (int i = 0; i < HS; ++i) vv[i] = vb[h * HS + i];
    const float* Wm = sW + (h * NR + r) * HS * HS;
    #pragma unroll
    for (int o = 0; o < HS; ++o) {
      float m = 0.f;
      #pragma unroll
      for (int i = 0; i < HS; ++i) m += vv[i] * Wm[i * HS + o];
      atomicAdd(&agg[(size_t)d * HID + h * HS + o], alpha * m);
    }
  }
}

// ------------- silu + gated residual + layernorm -----------------------------
__global__ void node_update(const float* __restrict__ y2, const float* __restrict__ xin,
                            const int* __restrict__ ntype, const float* __restrict__ skip_l,
                            const float* __restrict__ g, const float* __restrict__ b,
                            float* __restrict__ xout, int nRows) {
  int n = blockIdx.x;
  int c = threadIdx.x;                    // 0..127
  if (n >= nRows) return;
  __shared__ float red[HID];
  float y = y2[(size_t)n * HID + c];
  float o = y / (1.f + __expf(-y));       // silu
  float al = 1.f / (1.f + __expf(-skip_l[ntype[n]]));
  float v = o * al + xin[(size_t)n * HID + c] * (1.f - al);
  red[c] = v; __syncthreads();
  for (int s2 = 64; s2 > 0; s2 >>= 1) { if (c < s2) red[c] += red[c + s2]; __syncthreads(); }
  float mu = red[0] * (1.f / HID); __syncthreads();
  float dv = v - mu;
  red[c] = dv * dv; __syncthreads();
  for (int s2 = 64; s2 > 0; s2 >>= 1) { if (c < s2) red[c] += red[c + s2]; __syncthreads(); }
  float var = red[0] * (1.f / HID);
  xout[(size_t)n * HID + c] = dv * rsqrtf(var + LN_EPS) * g[c] + b[c];
}

// -----------------------------------------------------------------------------
extern "C" void kernel_launch(void* const* d_in, const int* in_sizes, int n_in,
                              void* d_out, int out_size, void* d_ws, size_t ws_size,
                              hipStream_t stream) {
  const float* x     = (const float*)d_in[0];
  const int*   edge  = (const int*)  d_in[1];
  const int*   ntype = (const int*)  d_in[2];
  const int*   etype = (const int*)  d_in[3];
  const float* Wk    = (const float*)d_in[4];
  const float* Wq    = (const float*)d_in[5];
  const float* Wv    = (const float*)d_in[6];
  const float* Wa    = (const float*)d_in[7];
  const float* pri   = (const float*)d_in[8];
  const float* Watt  = (const float*)d_in[9];
  const float* Wmsg  = (const float*)d_in[10];
  const float* skip  = (const float*)d_in[11];
  const float* ln_g  = (const float*)d_in[12];
  const float* ln_b  = (const float*)d_in[13];
  const float* Wff   = (const float*)d_in[14];
  const float* bff   = (const float*)d_in[15];

  const int Nn = in_sizes[0] / HID;
  const int Ee = in_sizes[1] / 2;
  const int tiles   = (Nn + 15) / 16;
  const int eblocks = (Ee + 255) / 256;

  float* ws = (float*)d_ws;
  size_t off = 0;
  float* e0   = ws + off; off += (size_t)Nn * HID;
  float* e1   = ws + off; off += (size_t)Nn * HID;
  float* kqv  = ws + off; off += (size_t)Nn * 384;
  float* abuf = ws + off; off += (size_t)Ee * NH;
  unsigned* nmax = (unsigned*)(ws + off); off += (size_t)Nn * NH;
  float* den  = ws + off; off += (size_t)Nn * NH;
  float* agg  = ws + off; off += (size_t)Nn * HID;
  float* y2   = ws + off; off += (size_t)Nn * HID;

  // bf16 swizzled weight region (64B aligned)
  size_t boff = (off * sizeof(float) + 63) & ~(size_t)63;
  __bf16* wswz = (__bf16*)((char*)d_ws + boff);
  const size_t typed_elems = (size_t)TN * 32 * 512;   // per typed matrix
  __bf16* swz[NLAYER][4];
  for (int l = 0; l < NLAYER; ++l)
    for (int m = 0; m < 4; ++m)
      swz[l][m] = wswz + ((size_t)l * 4 + m) * typed_elems;
  __bf16* swzff = wswz + (size_t)NLAYER * 4 * typed_elems;

  const size_t wsz = (size_t)TN * HID * HID;       // per-layer typed-weight slice
  const size_t rsz = (size_t)NH * NR * HS * HS;    // per-layer relation slice

  // one-time weight swizzle/convert (L2-resident afterwards)
  const float* mats[4] = { Wk, Wq, Wv, Wa };
  for (int l = 0; l < NLAYER; ++l)
    for (int m = 0; m < 4; ++m)
      swizzle_weights<<<TN * 32, 32, 0, stream>>>(mats[m] + l * wsz, swz[l][m]);
  swizzle_weights<<<32, 32, 0, stream>>>(Wff, swzff);

  const float* cur = x;
  float* nxt = e0;
  for (int l = 0; l < NLAYER; ++l) {
    (void)hipMemsetAsync(nmax, 0, (size_t)Nn * NH * sizeof(unsigned), stream);
    (void)hipMemsetAsync(den,  0, (size_t)Nn * NH * sizeof(float),    stream);
    (void)hipMemsetAsync(agg,  0, (size_t)Nn * HID * sizeof(float),   stream);

    typed_gemm128<<<tiles, 32, 0, stream>>>(cur, HID, ntype, swz[l][0], kqv,       384, Nn);
    typed_gemm128<<<tiles, 32, 0, stream>>>(cur, HID, ntype, swz[l][1], kqv + 128, 384, Nn);
    typed_gemm128<<<tiles, 32, 0, stream>>>(cur, HID, ntype, swz[l][2], kqv + 256, 384, Nn);

    edge_score<<<eblocks, 256, 0, stream>>>(kqv, edge, edge + Ee, etype,
        Watt + l * rsz, pri + (size_t)l * NH * NR, abuf, nmax, Ee);
    edge_exp<<<eblocks, 256, 0, stream>>>(abuf, edge + Ee, nmax, den, Ee);
    edge_msg<<<eblocks, 256, 0, stream>>>(kqv, edge, edge + Ee, etype,
        Wmsg + l * rsz, abuf, den, agg, Ee);

    typed_gemm128<<<tiles, 32, 0, stream>>>(agg, HID, ntype, swz[l][3], y2, HID, Nn);
    node_update<<<Nn, HID, 0, stream>>>(y2, cur, ntype, skip + l * TN,
                                        ln_g + l * HID, ln_b + l * HID, nxt, Nn);
    cur = nxt;
    nxt = (l == 0) ? e1 : e0;
  }
  gemm128_bias<<<tiles, 32, 0, stream>>>(cur, swzff, bff, (float*)d_out, Nn);
}
